// SingleFrameFlowTransformer_18236431139355
// MI455X (gfx1250) — matrix operand order repaired and checked by reading, not compiled, not run on hardware
//
#include <hip/hip_runtime.h>

// ---------------------------------------------------------------------------
// SingleFrameFlowTransformer for MI455X (gfx1250, wave32).
// - All dense GEMMs: v_wmma_f32_16x16x32_bf16 (bf16 in, fp32 acc), 128x64
//   macro-tiles, double-buffered LDS staging via GLOBAL_LOAD_ASYNC_TO_LDS_B128
//   (+ s_wait_asynccnt), 8 waves/block each computing a 32x32 block.
// - Dominant cost: per-layer cross-attn K/V projection (131072x1024x512),
//   824 GFLOP over 6 layers -> matrix-core bound; memory matrix in bf16.
// ---------------------------------------------------------------------------

typedef __bf16 bf16_t;
typedef __attribute__((ext_vector_type(16))) __bf16 bf16x16;
typedef __attribute__((ext_vector_type(8)))  __bf16 bf16x8;
typedef __attribute__((ext_vector_type(8)))  float  f32x8;

#define BB   32
#define JJ   27
#define JP   32      // J padded to WMMA tile multiple
#define EE   512
#define HH   8
#define DD   64
#define FFD  2048
#define NLAYER 6
#define NPT  4096

#define MBLK 128
#define NBLK 64
#define KSTEP 32

__constant__ int c_parent[JJ] = {0,0,1,2,3,4,5,6,7,8,8,3,11,12,13,14,15,15,0,18,19,20,0,22,23,24,3};
__constant__ float c_rscale[JJ] = {
  0.6f,0.6f,0.6f,0.6f, 1.0f,1.0f,1.0f,1.0f,1.0f, 1.5f,1.5f,
  1.0f,1.0f,1.0f,1.0f,1.0f, 1.5f,1.5f, 0.6f, 1.0f, 1.5f,1.5f,
  0.6f, 1.0f, 1.5f,1.5f,1.5f };

// ---------------------------------------------------------------------------
// CDNA5 async copy: global -> LDS, 16B per lane, tracked by ASYNCcnt.
// (CDNA5 ISA: GV mode, VDST holds LDS byte address.)
// ---------------------------------------------------------------------------
__device__ __forceinline__ void async_b128(unsigned lds_addr, const bf16_t* gaddr) {
  asm volatile("global_load_async_to_lds_b128 %0, %1, off"
               :: "v"(lds_addr), "v"(gaddr) : "memory");
}
__device__ __forceinline__ void wait_async3() {
  asm volatile("s_wait_asynccnt 0x3" ::: "memory");
}
__device__ __forceinline__ void wait_async0() {
  asm volatile("s_wait_asynccnt 0x0" ::: "memory");
}

// ---------------------------------------------------------------------------
// WMMA helpers (layouts per CDNA5 ISA §7.12.2)
// ---------------------------------------------------------------------------
__device__ __forceinline__ f32x8 wmma_bf16(bf16x16 a, bf16x16 b, f32x8 c) {
  return __builtin_amdgcn_wmma_f32_16x16x32_bf16(false, a, false, b, (short)0, c, false, false);
}

// A: 16x32 tile from row-major [M x K]; lane L<16: row L, K {0..7}U{16..23};
// lane L+16: row L, K {8..15}U{24..31}.
__device__ __forceinline__ bf16x16 load_fragA(const bf16_t* A, int lda, int lane) {
  int row = lane & 15;
  int kb  = (lane >> 4) * 8;
  const bf16_t* p = A + (long)row * lda + kb;
  bf16x8 lo = *(const bf16x8*)(p);       // K = kb..kb+7
  bf16x8 hi = *(const bf16x8*)(p + 16);  // K = kb+16..kb+23
  bf16x16 r;
#pragma unroll
  for (int i = 0; i < 8; i++) { r[i] = lo[i]; r[i + 8] = hi[i]; }
  return r;
}

// B: 32x16 tile, B[k,n] = W[n,k], W row-major [N x K].
// Lane L<16: col n=L, K {0..7}U{16..23}; lane L+16: col n=L, K {8..15}U{24..31}.
__device__ __forceinline__ bf16x16 load_fragB(const bf16_t* W, int ldw, int lane) {
  int col = lane & 15;
  int kb  = (lane >> 4) * 8;
  const bf16_t* p = W + (long)col * ldw + kb;
  bf16x8 lo = *(const bf16x8*)(p);
  bf16x8 hi = *(const bf16x8*)(p + 16);
  bf16x16 r;
#pragma unroll
  for (int i = 0; i < 8; i++) { r[i] = lo[i]; r[i + 8] = hi[i]; }
  return r;
}

// ---------------------------------------------------------------------------
// Macro-tiled WMMA GEMM: C[M,N] = A[M,K] @ W[N,K]^T (+bias)(+relu)(+residual)
// 256 threads = 8 waves (4 along M x 2 along N); each wave: 32x32 = 2x2 WMMAs.
// A(128x32) and B(64x32) K-slabs double-buffered in LDS via async copies.
// Requires M%128==0, N%64==0, K%32==0 (all shapes here comply).
// ---------------------------------------------------------------------------
__global__ void __launch_bounds__(256) gemm_wmma_lds(
    const bf16_t* __restrict__ A, int lda,
    const bf16_t* __restrict__ W, int ldw,
    const float* __restrict__ bias,
    float* __restrict__ outF, bf16_t* __restrict__ outB,
    const float* __restrict__ residual,
    int M, int Nn, int K, int flags /*1=relu*/) {
  __shared__ __align__(16) bf16_t sA[2][MBLK * KSTEP];  // 2 x 8 KB
  __shared__ __align__(16) bf16_t sB[2][NBLK * KSTEP];  // 2 x 4 KB
  int t = threadIdx.x;
  int lane = t & 31, wave = t >> 5;
  int wm = wave & 3, wn = wave >> 2;          // wave coords: 4 x 2
  int nblkN = Nn / NBLK;
  int bm = blockIdx.x / nblkN, bn = blockIdx.x % nblkN;
  const bf16_t* Ab = A + (long)bm * MBLK * lda;
  const bf16_t* Wb = W + (long)bn * NBLK * ldw;

  unsigned ldsA = (unsigned)(size_t)(&sA[0][0]);
  unsigned ldsB = (unsigned)(size_t)(&sB[0][0]);
  // per-thread copy slots: A has 128 rows x 4 chunks(16B) = 512; B has 256.
  int ca0 = t, ca1 = t + 256, cb = t;

  // issue one K-slab into LDS buffer `buf` (3 async b128 per thread)
  auto issue = [&](int buf, int k0) {
    async_b128(ldsA + buf * (MBLK * KSTEP * 2) + (ca0 >> 2) * 64 + (ca0 & 3) * 16,
               Ab + (long)(ca0 >> 2) * lda + k0 + (ca0 & 3) * 8);
    async_b128(ldsA + buf * (MBLK * KSTEP * 2) + (ca1 >> 2) * 64 + (ca1 & 3) * 16,
               Ab + (long)(ca1 >> 2) * lda + k0 + (ca1 & 3) * 8);
    async_b128(ldsB + buf * (NBLK * KSTEP * 2) + (cb >> 2) * 64 + (cb & 3) * 16,
               Wb + (long)(cb >> 2) * ldw + k0 + (cb & 3) * 8);
  };

  f32x8 acc00 = {}, acc01 = {}, acc10 = {}, acc11 = {};
  int nk = K / KSTEP;
  issue(0, 0);
  for (int ki = 0; ki < nk; ki++) {
    int buf = ki & 1;
    bool more = (ki + 1) < nk;
    if (more) issue(buf ^ 1, (ki + 1) * KSTEP);
    if (more) wait_async3(); else wait_async0();   // current slab landed (own wave)
    __syncthreads();                               // all waves' copies visible
    const bf16_t* a0 = &sA[buf][(wm * 32) * KSTEP];
    const bf16_t* b0 = &sB[buf][(wn * 32) * KSTEP];
    bf16x16 af0 = load_fragA(a0, KSTEP, lane);
    bf16x16 af1 = load_fragA(a0 + 16 * KSTEP, KSTEP, lane);
    bf16x16 bf0 = load_fragB(b0, KSTEP, lane);
    bf16x16 bf1 = load_fragB(b0 + 16 * KSTEP, KSTEP, lane);
    acc00 = wmma_bf16(af0, bf0, acc00);
    acc01 = wmma_bf16(af0, bf1, acc01);
    acc10 = wmma_bf16(af1, bf0, acc10);
    acc11 = wmma_bf16(af1, bf1, acc11);
    __syncthreads();                               // slab consumed; safe to refill
  }

  // epilogue: C/D layout -> row = base + r + 8*(lane>>4), col = base + (lane&15)
  int colb = bn * NBLK + wn * 32 + (lane & 15);
  int rowb = bm * MBLK + wm * 32 + ((lane >> 4) << 3);
#pragma unroll
  for (int sm = 0; sm < 2; sm++) {
#pragma unroll
    for (int sn = 0; sn < 2; sn++) {
      f32x8 acc = (sm == 0) ? (sn == 0 ? acc00 : acc01) : (sn == 0 ? acc10 : acc11);
      int col = colb + sn * 16;
      float bv = bias ? bias[col] : 0.f;
#pragma unroll
      for (int r = 0; r < 8; r++) {
        long idx = (long)(rowb + sm * 16 + r) * Nn + col;
        float v = acc[r] + bv;
        if (flags & 1) v = fmaxf(v, 0.f);
        if (residual) v += residual[idx];
        if (outF) outF[idx] = v;
        if (outB) outB[idx] = (bf16_t)v;
      }
    }
  }
}

// ---------------------------------------------------------------------------
// fp32 -> bf16 conversion
// ---------------------------------------------------------------------------
__global__ void k_f2b(const float* __restrict__ s, bf16_t* __restrict__ d, int n) {
  int i = blockIdx.x * 256 + threadIdx.x;
  if (i < n) d[i] = (bf16_t)s[i];
}

// ---------------------------------------------------------------------------
// Distances: per (b,j) mask bytes, argmin index, all-masked flag
// ---------------------------------------------------------------------------
__global__ void k_dist(const float* __restrict__ xt, const float* __restrict__ pc,
                       unsigned char* __restrict__ maskb, int* __restrict__ minidx,
                       int* __restrict__ allm) {
  int bj = blockIdx.x, b = bj / JJ, j = bj % JJ, t = threadIdx.x;
  float px = xt[bj * 3 + 0], py = xt[bj * 3 + 1], pz = xt[bj * 3 + 2];
  float rad = c_rscale[j] * 0.1f;
  __shared__ float sd[256]; __shared__ int sn[256];
  float bestd = 1e30f; int bestn = 0;
  for (int n = t; n < NPT; n += 256) {
    const float* p = pc + ((size_t)b * NPT + n) * 6;
    float dx = px - p[0], dy = py - p[1], dz = pz - p[2];
    float d2 = dx * dx + dy * dy + dz * dz;
    maskb[(size_t)bj * NPT + n] = (sqrtf(d2) > rad) ? 1 : 0;
    if (d2 < bestd || (d2 == bestd && n < bestn)) { bestd = d2; bestn = n; }
  }
  sd[t] = bestd; sn[t] = bestn; __syncthreads();
  for (int s = 128; s > 0; s >>= 1) {
    if (t < s) {
      if (sd[t + s] < sd[t] || (sd[t + s] == sd[t] && sn[t + s] < sn[t])) {
        sd[t] = sd[t + s]; sn[t] = sn[t + s];
      }
    }
    __syncthreads();
  }
  if (t == 0) { minidx[bj] = sn[0]; allm[bj] = (sqrtf(sd[0]) > rad) ? 1 : 0; }
}

// imp[b,n] = max_j exp(-d^2 / (2 sig_j^2)), sig_j = rad_j/2
__global__ void k_imp(const float* __restrict__ xt, const float* __restrict__ pc,
                      float* __restrict__ imp) {
  int idx = blockIdx.x * 256 + threadIdx.x;   // b*NPT + n
  int b = idx >> 12;
  const float* p = pc + (size_t)idx * 6;
  float x = p[0], y = p[1], z = p[2];
  float best = 0.f;
  for (int j = 0; j < JJ; j++) {
    const float* q = xt + ((size_t)b * JJ + j) * 3;
    float dx = q[0] - x, dy = q[1] - y, dz = q[2] - z;
    float d2 = dx * dx + dy * dy + dz * dz;
    float rad = c_rscale[j] * 0.1f, sig = rad * 0.5f;
    float dens = __expf(-d2 / (2.f * sig * sig));
    best = fmaxf(best, dens);
  }
  imp[idx] = best;
}

// memory[b,n,:] = (pc_raw @ Wpc^T + b) * imp  -> bf16
__global__ void k_mem(const float* __restrict__ pc, const float* __restrict__ imp,
                      const float* __restrict__ W, const float* __restrict__ bias,
                      bf16_t* __restrict__ memB) {
  int bn = blockIdx.x, t = threadIdx.x;
  __shared__ float sp[6]; __shared__ float si;
  if (t < 6) sp[t] = pc[(size_t)bn * 6 + t];
  if (t == 0) si = imp[bn];
  __syncthreads();
  for (int e = t; e < EE; e += 256) {
    const float* w = W + (size_t)e * 6;
    float acc = bias[e];
    for (int k = 0; k < 6; k++) acc += sp[k] * w[k];
    memB[(size_t)bn * EE + e] = (bf16_t)(acc * si);
  }
}

// h_dop[b,:] = mean_n dop2(silu(dop1(dop)))
__global__ void k_hdop(const float* __restrict__ pc, const float* __restrict__ w1,
                       const float* __restrict__ b1, const float* __restrict__ w2,
                       const float* __restrict__ b2v, float* __restrict__ hdop) {
  int b = blockIdx.x, t = threadIdx.x;
  __shared__ float h1[4][DD];
  float acc0 = 0.f, acc1 = 0.f;
  for (int n0 = 0; n0 < NPT; n0 += 4) {
    int q = t >> 6, i = t & 63;
    float dop = pc[((size_t)b * NPT + n0 + q) * 6 + 3];
    float u = w1[i] * dop + b1[i];
    h1[q][i] = u / (1.f + __expf(-u));
    __syncthreads();
    for (int qq = 0; qq < 4; qq++) {
      const float* h  = h1[qq];
      const float* W0 = w2 + (size_t)t * DD;
      const float* W1 = w2 + (size_t)(t + 256) * DD;
      float a0 = 0.f, a1 = 0.f;
      for (int i2 = 0; i2 < DD; i2++) { a0 += W0[i2] * h[i2]; a1 += W1[i2] * h[i2]; }
      acc0 += a0; acc1 += a1;
    }
    __syncthreads();
  }
  float invN = 1.f / (float)NPT;
  hdop[(size_t)b * EE + t]       = acc0 * invN + b2v[t];
  hdop[(size_t)b * EE + t + 256] = acc1 * invN + b2v[t + 256];
}

// h_tau[b,:] = time2(silu(time1(tau)))
__global__ void k_htau(const float* __restrict__ tau, const float* __restrict__ w1,
                       const float* __restrict__ b1, const float* __restrict__ w2,
                       const float* __restrict__ b2v, float* __restrict__ htau) {
  int b = blockIdx.x, t = threadIdx.x;
  __shared__ float t1[EE];
  float tv = tau[b];
  for (int i = t; i < EE; i += 256) { float u = w1[i] * tv + b1[i]; t1[i] = u / (1.f + __expf(-u)); }
  __syncthreads();
  for (int e = t; e < EE; e += 256) {
    float acc = b2v[e];
    const float* W = w2 + (size_t)e * EE;
    for (int i = 0; i < EE; i++) acc += W[i] * t1[i];
    htau[(size_t)b * EE + e] = acc;
  }
}

// h_xt[b,j,:] = joint_embed(x_t)
__global__ void k_hxt(const float* __restrict__ xt, const float* __restrict__ w,
                      const float* __restrict__ bi, float* __restrict__ hxt) {
  int bj = blockIdx.x, t = threadIdx.x;
  float x0 = xt[bj * 3], x1 = xt[bj * 3 + 1], x2 = xt[bj * 3 + 2];
  for (int e = t; e < EE; e += 256)
    hxt[(size_t)bj * EE + e] = bi[e] + x0 * w[e * 3] + x1 * w[e * 3 + 1] + x2 * w[e * 3 + 2];
}

// query = diff_proj([h_xt, h_parent]) + h_tau + coarse_embed + diff_embed + id + h_dop
__global__ void k_query(const float* __restrict__ xt, const float* __restrict__ xc,
                        const float* __restrict__ hxt, const float* __restrict__ htau,
                        const float* __restrict__ hdop,
                        const float* __restrict__ cew, const float* __restrict__ ceb,
                        const float* __restrict__ dew, const float* __restrict__ deb,
                        const float* __restrict__ dpw, const float* __restrict__ dpb,
                        const float* __restrict__ jid, float* __restrict__ x) {
  int row = blockIdx.x, b = row / JP, j = row % JP, t = threadIdx.x;
  if (j >= JJ) { x[(size_t)row * EE + t] = 0.f; x[(size_t)row * EE + t + 256] = 0.f; return; }
  int par = c_parent[j];
  const float* hj = hxt + ((size_t)b * JJ + j) * EE;
  const float* hp = hxt + ((size_t)b * JJ + par) * EE;
  int bj = b * JJ + j;
  float c0 = xc[bj * 3], c1 = xc[bj * 3 + 1], c2 = xc[bj * 3 + 2];
  float d0 = xt[bj * 3] - c0, d1 = xt[bj * 3 + 1] - c1, d2 = xt[bj * 3 + 2] - c2;
  for (int e = t; e < EE; e += 256) {
    const float* W = dpw + (size_t)e * (2 * EE);
    float acc = dpb[e];
    for (int k = 0; k < EE; k++) acc += hj[k] * W[k];
    for (int k = 0; k < EE; k++) acc += hp[k] * W[EE + k];
    acc += ceb[e] + c0 * cew[e * 3] + c1 * cew[e * 3 + 1] + c2 * cew[e * 3 + 2];
    acc += deb[e] + d0 * dew[e * 3] + d1 * dew[e * 3 + 1] + d2 * dew[e * 3 + 2];
    acc += jid[(size_t)j * EE + e] + htau[(size_t)b * EE + e] + hdop[(size_t)b * EE + e];
    x[(size_t)row * EE + e] = acc;
  }
}

// LayerNorm over E=512, bf16 output; zero-fills padded rows j>=27.
__global__ void k_ln(const float* __restrict__ x, const float* __restrict__ g,
                     const float* __restrict__ be, bf16_t* __restrict__ out) {
  int row = blockIdx.x, j = row % JP, t = threadIdx.x;
  if (j >= JJ) {
    out[(size_t)row * EE + t] = (bf16_t)0.f;
    out[(size_t)row * EE + t + 256] = (bf16_t)0.f;
    return;
  }
  __shared__ float s1[256], s2[256];
  float a = x[(size_t)row * EE + t], b2 = x[(size_t)row * EE + t + 256];
  s1[t] = a + b2; s2[t] = a * a + b2 * b2;
  __syncthreads();
  for (int s = 128; s > 0; s >>= 1) {
    if (t < s) { s1[t] += s1[t + s]; s2[t] += s2[t + s]; }
    __syncthreads();
  }
  float mu = s1[0] * (1.f / EE);
  float var = s2[0] * (1.f / EE) - mu * mu;
  float rs = rsqrtf(var + 1e-5f);
  out[(size_t)row * EE + t]       = (bf16_t)((a  - mu) * rs * g[t] + be[t]);
  out[(size_t)row * EE + t + 256] = (bf16_t)((b2 - mu) * rs * g[t + 256] + be[t + 256]);
}

// Self-attention over 27 tokens with topology bias; one block per (b,h).
__global__ void k_selfattn(const bf16_t* __restrict__ qkv, bf16_t* __restrict__ attnout) {
  int b = blockIdx.x / HH, h = blockIdx.x % HH, t = threadIdx.x;
  __shared__ float S[JJ][JJ + 1];
  const bf16_t* base = qkv + (size_t)b * JP * (3 * EE);
  for (int p = t; p < JJ * JJ; p += 256) {
    int q = p / JJ, k = p % JJ;
    const bf16_t* Qp = base + (size_t)q * (3 * EE) + h * DD;
    const bf16_t* Kp = base + (size_t)k * (3 * EE) + EE + h * DD;
    float s = 0.f;
#pragma unroll 8
    for (int d = 0; d < DD; d++) s += (float)Qp[d] * (float)Kp[d];
    int pq = c_parent[q], pk = c_parent[k];
    float bias = (q == k || pq == k || pk == q) ? 0.f : -1e30f;
    S[q][k] = s * 0.125f + bias;
  }
  __syncthreads();
  if (t < JJ) {
    float m = -1e30f;
    for (int k = 0; k < JJ; k++) m = fmaxf(m, S[t][k]);
    float sum = 0.f;
    for (int k = 0; k < JJ; k++) { float p = __expf(S[t][k] - m); S[t][k] = p; sum += p; }
    float inv = 1.f / sum;
    for (int k = 0; k < JJ; k++) S[t][k] *= inv;
  }
  __syncthreads();
  for (int idx = t; idx < JP * DD; idx += 256) {
    int q = idx / DD, d = idx % DD;
    float o = 0.f;
    if (q < JJ)
      for (int k = 0; k < JJ; k++)
        o += S[q][k] * (float)base[(size_t)k * (3 * EE) + 2 * EE + h * DD + d];
    attnout[((size_t)b * JP + q) * EE + h * DD + d] = (bf16_t)o;
  }
}

// Cross-attention: 27 queries vs 4096 keys with radius/rescue mask.
// One block per (b,h); scores buffered in LDS (16KB), two-pass softmax.
__global__ void k_crossattn(const bf16_t* __restrict__ qbuf, const bf16_t* __restrict__ kv,
                            const unsigned char* __restrict__ maskb,
                            const int* __restrict__ minidx, const int* __restrict__ allm,
                            bf16_t* __restrict__ attnout) {
  int b = blockIdx.x / HH, h = blockIdx.x % HH, t = threadIdx.x;
  __shared__ float sc[NPT];
  __shared__ float red[256];
  __shared__ float qsh[DD];
  __shared__ float op[4][DD];
  const bf16_t* kvb = kv + (size_t)b * NPT * (2 * EE);
  for (int q = 0; q < JJ; q++) {
    if (t < DD) qsh[t] = (float)qbuf[((size_t)b * JP + q) * EE + h * DD + t];
    __syncthreads();
    const unsigned char* mrow = maskb + ((size_t)b * JJ + q) * NPT;
    int am = allm[b * JJ + q], mi = minidx[b * JJ + q];
    float lm = -1e30f;
    for (int n = t; n < NPT; n += 256) {
      const bf16_t* kp = kvb + (size_t)n * (2 * EE) + h * DD;
      __builtin_prefetch(kvb + (size_t)(n + 256) * (2 * EE) + h * DD, 0, 1);
      float s = 0.f;
#pragma unroll 8
      for (int d = 0; d < DD; d++) s += qsh[d] * (float)kp[d];
      s *= 0.125f;
      int masked = am ? (n != mi) : (int)mrow[n];
      if (masked) s = -1e30f;
      sc[n] = s; lm = fmaxf(lm, s);
    }
    red[t] = lm; __syncthreads();
    for (int s = 128; s > 0; s >>= 1) { if (t < s) red[t] = fmaxf(red[t], red[t + s]); __syncthreads(); }
    float m = red[0]; __syncthreads();
    float ls = 0.f;
    for (int n = t; n < NPT; n += 256) { float p = __expf(sc[n] - m); sc[n] = p; ls += p; }
    red[t] = ls; __syncthreads();
    for (int s = 128; s > 0; s >>= 1) { if (t < s) red[t] += red[t + s]; __syncthreads(); }
    float inv = 1.f / red[0];
    __syncthreads();
    int d = t & 63, part = t >> 6;
    float acc = 0.f;
    for (int n = part; n < NPT; n += 4)
      acc += sc[n] * (float)kvb[(size_t)n * (2 * EE) + EE + h * DD + d];
    op[part][d] = acc; __syncthreads();
    if (t < DD) {
      float o = (op[0][t] + op[1][t] + op[2][t] + op[3][t]) * inv;
      attnout[((size_t)b * JP + q) * EE + h * DD + t] = (bf16_t)o;
    }
    __syncthreads();
  }
  for (int idx = t; idx < (JP - JJ) * DD; idx += 256) {
    int q = JJ + idx / DD, d = idx % DD;
    attnout[((size_t)b * JP + q) * EE + h * DD + d] = (bf16_t)0.f;
  }
}

// Final projection to velocities (E->3)
__global__ void k_out(const float* __restrict__ x, const float* __restrict__ vw,
                      const float* __restrict__ vb, float* __restrict__ out) {
  int bj = blockIdx.x, b = bj / JJ, j = bj % JJ, t = threadIdx.x;
  if (t < 3) {
    const float* xr = x + ((size_t)b * JP + j) * EE;
    const float* W = vw + (size_t)t * EE;
    float acc = vb[t];
    for (int e = 0; e < EE; e++) acc += xr[e] * W[e];
    out[(size_t)bj * 3 + t] = acc;
  }
}

// ---------------------------------------------------------------------------
// Host orchestration
// ---------------------------------------------------------------------------
enum {
  IN_XT = 0, IN_TAU, IN_XC, IN_PC,
  P_PCW, P_PCB, P_JEW, P_JEB, P_CEW, P_CEB, P_DEW, P_DEB,
  P_D1W, P_D1B, P_D2W, P_D2B, P_T1W, P_T1B, P_T2W, P_T2B,
  P_DPW, P_DPB, P_VW, P_VB, P_JID, P_LAYER0
};
// per-layer param stride 18:
// +0 sa_in_w +1 sa_in_b +2 sa_out_w +3 sa_out_b +4 ca_in_w +5 ca_in_b
// +6 ca_out_w +7 ca_out_b +8 ff1_w +9 ff1_b +10 ff2_w +11 ff2_b
// +12 ln1_g +13 ln1_b +14 ln2_g +15 ln2_b +16 ln3_g +17 ln3_b

static void launch_gemm(const bf16_t* A, int lda, const bf16_t* W, int ldw,
                        const float* bias, float* outF, bf16_t* outB, const float* res,
                        int M, int Nn, int K, int flags, hipStream_t s) {
  int blocks = (M / MBLK) * (Nn / NBLK);
  gemm_wmma_lds<<<blocks, 256, 0, s>>>(A, lda, W, ldw, bias, outF, outB, res, M, Nn, K, flags);
}

extern "C" void kernel_launch(void* const* d_in, const int* in_sizes, int n_in,
                              void* d_out, int out_size, void* d_ws, size_t ws_size,
                              hipStream_t stream) {
  auto P = [&](int i) -> const float* { return (const float*)d_in[i]; };
  const float* xt  = P(IN_XT);
  const float* tau = P(IN_TAU);
  const float* xc  = P(IN_XC);
  const float* pc  = P(IN_PC);

  // workspace bump allocator (256B aligned)
  char* wsb = (char*)d_ws;
  size_t off = 0;
  auto alloc = [&](size_t bytes) -> void* {
    off = (off + 255) & ~(size_t)255;
    void* p = wsb + off; off += bytes; return p;
  };
  bf16_t* memB    = (bf16_t*)alloc((size_t)BB * NPT * EE * 2);
  bf16_t* kvB     = (bf16_t*)alloc((size_t)BB * NPT * (2 * EE) * 2);
  float*  imp     = (float*)alloc((size_t)BB * NPT * 4);
  unsigned char* maskb = (unsigned char*)alloc((size_t)BB * JJ * NPT);
  int*    minidx  = (int*)alloc((size_t)BB * JJ * 4);
  int*    allm    = (int*)alloc((size_t)BB * JJ * 4);
  float*  hdop    = (float*)alloc((size_t)BB * EE * 4);
  float*  htau    = (float*)alloc((size_t)BB * EE * 4);
  float*  hxt     = (float*)alloc((size_t)BB * JJ * EE * 4);
  float*  xbuf    = (float*)alloc((size_t)BB * JP * EE * 4);
  bf16_t* hln     = (bf16_t*)alloc((size_t)BB * JP * EE * 2);
  bf16_t* qkv     = (bf16_t*)alloc((size_t)BB * JP * (3 * EE) * 2);
  bf16_t* attnout = (bf16_t*)alloc((size_t)BB * JP * EE * 2);
  bf16_t* ffmid   = (bf16_t*)alloc((size_t)BB * JP * FFD * 2);
  // per-layer bf16 weight block: sa_in, sa_out, ca_in, ca_out, ff1, ff2
  const size_t LW = (size_t)1536 * 512 + 512 * 512 + (size_t)1536 * 512 + 512 * 512 +
                    (size_t)2048 * 512 + (size_t)512 * 2048;  // 4,194,304 elems
  bf16_t* wbf = (bf16_t*)alloc((size_t)NLAYER * LW * 2);

  auto conv = [&](const float* src, bf16_t* dst, size_t n) {
    k_f2b<<<(int)((n + 255) / 256), 256, 0, stream>>>(src, dst, (int)n);
  };

  // ---- preprocessing ----
  k_dist<<<BB * JJ, 256, 0, stream>>>(xt, pc, maskb, minidx, allm);
  k_imp<<<BB * NPT / 256, 256, 0, stream>>>(xt, pc, imp);
  k_mem<<<BB * NPT, 256, 0, stream>>>(pc, imp, P(P_PCW), P(P_PCB), memB);
  k_hdop<<<BB, 256, 0, stream>>>(pc, P(P_D1W), P(P_D1B), P(P_D2W), P(P_D2B), hdop);
  k_htau<<<BB, 256, 0, stream>>>(tau, P(P_T1W), P(P_T1B), P(P_T2W), P(P_T2B), htau);
  k_hxt<<<BB * JJ, 256, 0, stream>>>(xt, P(P_JEW), P(P_JEB), hxt);
  k_query<<<BB * JP, 256, 0, stream>>>(xt, xc, hxt, htau, hdop,
                                       P(P_CEW), P(P_CEB), P(P_DEW), P(P_DEB),
                                       P(P_DPW), P(P_DPB), P(P_JID), xbuf);

  // ---- weight conversion to bf16 (once) ----
  for (int L = 0; L < NLAYER; L++) {
    int LP = P_LAYER0 + L * 18;
    bf16_t* base   = wbf + (size_t)L * LW;
    bf16_t* sa_in  = base;
    bf16_t* sa_out = base + (size_t)1536 * 512;
    bf16_t* ca_in  = sa_out + (size_t)512 * 512;
    bf16_t* ca_out = ca_in + (size_t)1536 * 512;
    bf16_t* ff1    = ca_out + (size_t)512 * 512;
    bf16_t* ff2    = ff1 + (size_t)2048 * 512;
    conv(P(LP + 0),  sa_in,  (size_t)1536 * 512);
    conv(P(LP + 2),  sa_out, (size_t)512 * 512);
    conv(P(LP + 4),  ca_in,  (size_t)1536 * 512);
    conv(P(LP + 6),  ca_out, (size_t)512 * 512);
    conv(P(LP + 8),  ff1,    (size_t)2048 * 512);
    conv(P(LP + 10), ff2,    (size_t)512 * 2048);
  }

  // ---- transformer layers ----
  const int Mx = BB * JP;  // 1024 padded rows
  for (int L = 0; L < NLAYER; L++) {
    int LP = P_LAYER0 + L * 18;
    bf16_t* base   = wbf + (size_t)L * LW;
    bf16_t* sa_in  = base;
    bf16_t* sa_out = base + (size_t)1536 * 512;
    bf16_t* ca_in  = sa_out + (size_t)512 * 512;
    bf16_t* ca_out = ca_in + (size_t)1536 * 512;
    bf16_t* ff1    = ca_out + (size_t)512 * 512;
    bf16_t* ff2    = ff1 + (size_t)2048 * 512;

    // self-attention block
    k_ln<<<Mx, 256, 0, stream>>>(xbuf, P(LP + 12), P(LP + 13), hln);
    launch_gemm(hln, 512, sa_in, 512, P(LP + 1), nullptr, qkv, nullptr, Mx, 1536, 512, 0, stream);
    k_selfattn<<<BB * HH, 256, 0, stream>>>(qkv, attnout);
    launch_gemm(attnout, 512, sa_out, 512, P(LP + 3), xbuf, nullptr, xbuf, Mx, 512, 512, 0, stream);

    // cross-attention block
    k_ln<<<Mx, 256, 0, stream>>>(xbuf, P(LP + 14), P(LP + 15), hln);
    launch_gemm(hln, 512, ca_in, 512, P(LP + 5), nullptr, qkv, nullptr, Mx, 512, 512, 0, stream);
    launch_gemm(memB, 512, ca_in + (size_t)512 * 512, 512, P(LP + 5) + 512,
                nullptr, kvB, nullptr, BB * NPT, 2 * EE, 512, 0, stream);  // the big one
    k_crossattn<<<BB * HH, 256, 0, stream>>>(qkv, kvB, maskb, minidx, allm, attnout);
    launch_gemm(attnout, 512, ca_out, 512, P(LP + 7), xbuf, nullptr, xbuf, Mx, 512, 512, 0, stream);

    // feed-forward block
    k_ln<<<Mx, 256, 0, stream>>>(xbuf, P(LP + 16), P(LP + 17), hln);
    launch_gemm(hln, 512, ff1, 512, P(LP + 9), nullptr, ffmid, nullptr, Mx, FFD, 512, 1, stream);
    launch_gemm(ffmid, FFD, ff2, FFD, P(LP + 11), xbuf, nullptr, xbuf, Mx, 512, FFD, 0, stream);
  }

  k_out<<<BB * JJ, 64, 0, stream>>>(xbuf, P(P_VW), P(P_VB), (float*)d_out);
}
